// BHCH_33500744908985
// MI455X (gfx1250) — matrix operand order repaired and checked by reading, not compiled, not run on hardware
//
#include <hip/hip_runtime.h>
#include <math.h>

typedef _Float16 h16;
typedef __attribute__((ext_vector_type(4)))  _Float16 v4h;
typedef __attribute__((ext_vector_type(8)))  _Float16 v8h;
typedef __attribute__((ext_vector_type(16))) _Float16 v16h;
typedef __attribute__((ext_vector_type(8)))  float    v8f;

// ---------------- workspace layout (float elements) ----------------
constexpr size_t O_X     = 0;                    // act_x  (2,2048,512)
constexpr size_t O_SHORT = O_X     + 2097152;    // short  (2,2048,512)
constexpr size_t O_MOUT  = O_SHORT + 2097152;    // mamba-sum (2,2048,512)
constexpr size_t O_RES   = O_MOUT  + 2097152;    // res (2048,1024)
constexpr size_t O_M     = O_RES   + 2097152;    // m (2,2048,512)
constexpr size_t O_S     = O_M     + 2097152;    // scratch base
// mamba scratch
constexpr size_t O_XF    = O_S;                  // (2,2048,512) x_flat / "out"
constexpr size_t O_XZ    = O_XF    + 2097152;    // (2,2048,2048) xz / generic temp
constexpr size_t O_XI    = O_XZ    + 8388608;    // (2,2048,1024)
constexpr size_t O_DT    = O_XI    + 4194304;    // (2,2048,1024) ; also mix buffer
constexpr size_t O_DBC   = O_DT    + 4194304;    // (2,2048,64)
constexpr size_t O_Y     = O_DBC   + 262144;     // (2,2048,1024)
constexpr size_t O_R     = O_Y     + 4194304;    // 4096 cos-sims
// attention scratch (aliases mamba scratch, used strictly after)
constexpr size_t O_QKV   = O_S;                  // (2048,3072)
constexpr size_t O_SC    = O_QKV   + 6291456;    // (2048,2048) per-head scores
constexpr size_t O_AO    = O_SC    + 4194304;    // (2048,1024)
constexpr size_t O_XLN   = O_AO    + 2097152;    // (2048,1024)
constexpr size_t O_FF    = O_XLN   + 2097152;    // (2048,2048)
constexpr size_t O_ENC   = O_FF    + 4194304;    // (2048,1024)

// ---------------- generic WMMA f16-in / f32-acc GEMM ----------------
// MODE: 0=plain, 1=transB, 2=flip B K-index, 3=flip B N-index
// epi:  0=none 1=bias 2=bias+silu 3=bias+softplus 4=bias+relu 5=bias+tanh
// REQUIRES: M%128==0, N%64==0, K%32==0 (all call sites satisfy this)
template <int MODE>
__launch_bounds__(256)
__global__ void gemm_f16_wmma(const float* __restrict__ A, int lda,
                              const float* __restrict__ B, int ldb,
                              float* __restrict__ C, int ldc,
                              const float* __restrict__ bias,
                              int M, int N, int K,
                              float alpha, int accC, int epi)
{
  __shared__ h16 As[128 * 32];   // M x K tile
  __shared__ h16 Bs[64 * 32];    // N x K tile (transposed in LDS)
  const int tid  = threadIdx.x;
  const int lane = tid & 31;
  const int wave = tid >> 5;     // 8 waves
  const int wm   = wave >> 1;    // 0..3
  const int wn   = wave & 1;     // 0..1
  const int m0   = blockIdx.y * 128;
  const int n0   = blockIdx.x * 64;
  const int half = lane >> 4;
  const int l16  = lane & 15;

  float4 pa[4];
  float4 pb[2];

  auto loadA = [&](int k0) {
#pragma unroll
    for (int j = 0; j < 4; ++j) {
      int f = j * 256 + tid;             // float4 index over 128x32 tile
      int r = f >> 3, kq = (f & 7) * 4;
      pa[j] = *reinterpret_cast<const float4*>(&A[(size_t)(m0 + r) * lda + k0 + kq]);
    }
  };
  auto loadB = [&](int k0) {
#pragma unroll
    for (int j = 0; j < 2; ++j) {
      int f = j * 256 + tid;             // float4 index over 32x64 tile
      if (MODE == 1) {                   // transB: contiguous along K
        int nn = f >> 3, kq = (f & 7) * 4;
        pb[j] = *reinterpret_cast<const float4*>(&B[(size_t)(n0 + nn) * ldb + k0 + kq]);
      } else {                           // row-major B: contiguous along N
        int kk = f >> 4, nq = (f & 15) * 4;
        int sk = (MODE == 2) ? (K - 1 - (k0 + kk)) : (k0 + kk);
        if (MODE == 3) {
          float4 t = *reinterpret_cast<const float4*>(&B[(size_t)sk * ldb + (N - 4 - (n0 + nq))]);
          pb[j].x = t.w; pb[j].y = t.z; pb[j].z = t.y; pb[j].w = t.x;
        } else {
          pb[j] = *reinterpret_cast<const float4*>(&B[(size_t)sk * ldb + n0 + nq]);
        }
      }
    }
  };
  auto storeA = [&]() {
#pragma unroll
    for (int j = 0; j < 4; ++j) {
      int f = j * 256 + tid;
      int r = f >> 3, kq = (f & 7) * 4;
      v4h h; h[0] = (h16)pa[j].x; h[1] = (h16)pa[j].y; h[2] = (h16)pa[j].z; h[3] = (h16)pa[j].w;
      *reinterpret_cast<v4h*>(&As[r * 32 + kq]) = h;
    }
  };
  auto storeB = [&]() {
#pragma unroll
    for (int j = 0; j < 2; ++j) {
      int f = j * 256 + tid;
      if (MODE == 1) {
        int nn = f >> 3, kq = (f & 7) * 4;
        v4h h; h[0] = (h16)pb[j].x; h[1] = (h16)pb[j].y; h[2] = (h16)pb[j].z; h[3] = (h16)pb[j].w;
        *reinterpret_cast<v4h*>(&Bs[nn * 32 + kq]) = h;
      } else {
        int kk = f >> 4, nq = (f & 15) * 4;
        Bs[(nq + 0) * 32 + kk] = (h16)pb[j].x;
        Bs[(nq + 1) * 32 + kk] = (h16)pb[j].y;
        Bs[(nq + 2) * 32 + kk] = (h16)pb[j].z;
        Bs[(nq + 3) * 32 + kk] = (h16)pb[j].w;
      }
    }
  };

  v8f acc[2][2] = {};

  loadA(0); loadB(0);
  storeA(); storeB();
  __syncthreads();

  for (int k0 = 0; k0 < K; k0 += 32) {
    const bool more = (k0 + 32) < K;
    if (more) {                          // register prefetch of next tile
      loadA(k0 + 32);
      loadB(k0 + 32);
      if (k0 + 64 < K)                   // L2 prefetch of the tile after that
        __builtin_prefetch(&A[(size_t)(m0 + (tid >> 1)) * lda + k0 + 64], 0, 1);
    }

    v16h afr[2], bfr[2];
#pragma unroll
    for (int t = 0; t < 2; ++t) {
      int row = wm * 32 + t * 16 + l16;  // A: lane holds row; K split lo/hi halves
      v8h alo = *reinterpret_cast<const v8h*>(&As[row * 32 + half * 8]);
      v8h ahi = *reinterpret_cast<const v8h*>(&As[row * 32 + 16 + half * 8]);
      afr[t] = __builtin_shufflevector(alo, ahi, 0,1,2,3,4,5,6,7,8,9,10,11,12,13,14,15);
      int col = wn * 32 + t * 16 + l16;  // B: lane holds column; K split lo/hi halves
      v8h blo = *reinterpret_cast<const v8h*>(&Bs[col * 32 + half * 8]);
      v8h bhi = *reinterpret_cast<const v8h*>(&Bs[col * 32 + 16 + half * 8]);
      bfr[t] = __builtin_shufflevector(blo, bhi, 0,1,2,3,4,5,6,7,8,9,10,11,12,13,14,15);
    }
#pragma unroll
    for (int tm = 0; tm < 2; ++tm)
#pragma unroll
      for (int tn = 0; tn < 2; ++tn)
        acc[tm][tn] = __builtin_amdgcn_wmma_f32_16x16x32_f16(
            false, afr[tm], false, bfr[tn], (short)0, acc[tm][tn], false, false);
    __syncthreads();

    if (more) {
      storeA(); storeB();
      __syncthreads();
    }
  }

  // ---- epilogue: phased so loads batch and activation branches once ----
  const int gmB = m0 + wm * 32 + half * 8;     // + tm*16 + r
  const int gnB = n0 + wn * 32 + l16;          // + tn*16
  float bval[2] = {0.f, 0.f};
  if (bias) { bval[0] = bias[gnB]; bval[1] = bias[gnB + 16]; }

  float vout[2][2][8];
#pragma unroll
  for (int tm = 0; tm < 2; ++tm)
#pragma unroll
    for (int tn = 0; tn < 2; ++tn)
#pragma unroll
      for (int r = 0; r < 8; ++r)
        vout[tm][tn][r] = alpha * acc[tm][tn][r] + bval[tn];

  float* vp = &vout[0][0][0];
  if (epi == 2)      { for (int i = 0; i < 32; ++i) { float v = vp[i]; vp[i] = v / (1.f + expf(-v)); } }
  else if (epi == 3) { for (int i = 0; i < 32; ++i) { float v = vp[i]; vp[i] = (v > 20.f) ? v : log1pf(expf(v)); } }
  else if (epi == 4) { for (int i = 0; i < 32; ++i) vp[i] = fmaxf(vp[i], 0.f); }
  else if (epi == 5) { for (int i = 0; i < 32; ++i) vp[i] = tanhf(vp[i]); }

  if (accC) {
    float cold[2][2][8];
#pragma unroll
    for (int tm = 0; tm < 2; ++tm)
#pragma unroll
      for (int tn = 0; tn < 2; ++tn)
#pragma unroll
        for (int r = 0; r < 8; ++r)
          cold[tm][tn][r] = C[(size_t)(gmB + tm * 16 + r) * ldc + gnB + tn * 16];
#pragma unroll
    for (int tm = 0; tm < 2; ++tm)
#pragma unroll
      for (int tn = 0; tn < 2; ++tn)
#pragma unroll
        for (int r = 0; r < 8; ++r)
          vout[tm][tn][r] += cold[tm][tn][r];
  }

#pragma unroll
  for (int tm = 0; tm < 2; ++tm)
#pragma unroll
    for (int tn = 0; tn < 2; ++tn)
#pragma unroll
      for (int r = 0; r < 8; ++r)
        C[(size_t)(gmB + tm * 16 + r) * ldc + gnB + tn * 16] = vout[tm][tn][r];
}

// ---------------- elementwise / reduction kernels ----------------
__global__ void fill_zero(float* p, size_t n) {
  size_t i = (size_t)blockIdx.x * 256 + threadIdx.x;
  if (i < n) p[i] = 0.f;
}

// out[c*rows + r] = in[r*cols + c] * scale
__global__ void transpose_scale(const float* __restrict__ in, float* __restrict__ out,
                                int rows, int cols, const float* __restrict__ scale)
{
  __shared__ float tile[32][33];
  int bx = blockIdx.x * 32, by = blockIdx.y * 32;
  int tx = threadIdx.x & 31, ty = threadIdx.x >> 5;
  float sc = scale ? *scale : 1.f;
  for (int j = 0; j < 32; j += 8) {
    int r = by + ty + j, c = bx + tx;
    tile[ty + j][tx] = (r < rows && c < cols) ? in[(size_t)r * cols + c] : 0.f;
  }
  __syncthreads();
  for (int j = 0; j < 32; j += 8) {
    int r = bx + ty + j, c = by + tx;
    if (r < cols && c < rows) out[(size_t)r * rows + c] = tile[tx][ty + j] * sc;
  }
}

__launch_bounds__(256)
__global__ void grn_kernel(const float* __restrict__ in, float* __restrict__ out,
                           const float* __restrict__ g, const float* __restrict__ b,
                           int C, int do_silu)
{
  __shared__ float red[256];
  int row = blockIdx.x;
  const float* x = in + (size_t)row * C;
  float* o = out + (size_t)row * C;
  float ss = 0.f;
  for (int c = threadIdx.x; c < C; c += 256) { float v = x[c]; ss += v * v; }
  red[threadIdx.x] = ss; __syncthreads();
  for (int st = 128; st; st >>= 1) {
    if (threadIdx.x < st) red[threadIdx.x] += red[threadIdx.x + st];
    __syncthreads();
  }
  float n  = sqrtf(red[0]);
  float Nx = n / (n + 1e-6f);
  float iv = 1.f / fmaxf(n, 1e-12f);
  for (int c = threadIdx.x; c < C; c += 256) {
    float v = x[c];
    float y = g[c] * v * Nx + b[c] + v * iv;
    if (do_silu) y = y / (1.f + expf(-y));
    o[c] = y;
  }
}

// depthwise causal conv (k=4) + bias + silu; rev flips direction
__global__ void conv_silu(const float* __restrict__ xz, const float* __restrict__ w,
                          const float* __restrict__ b, float* __restrict__ xi, int rev)
{
  int i = blockIdx.x * 256 + threadIdx.x;
  if (i >= 2 * 2048 * 1024) return;
  int c = i & 1023;
  int l = (i >> 10) & 2047;
  int s = i >> 21;
  const float* X = xz + (size_t)s * 2048 * 2048;
  float acc = b[c];
#pragma unroll
  for (int k = 0; k < 4; ++k) {
    int ll = rev ? (l + 3 - k) : (l - 3 + k);
    if (ll >= 0 && ll < 2048) acc += w[c * 4 + k] * X[(size_t)ll * 2048 + c];
  }
  xi[i] = acc / (1.f + expf(-acc));
}

// selective scan: 1 thread per (s, channel); 16-wide state; next step prefetched
__launch_bounds__(256)
__global__ void mamba_scan(const float* __restrict__ dt, const float* __restrict__ xi,
                           const float* __restrict__ dbc, const float* __restrict__ Alog,
                           float* __restrict__ y, int rev)
{
  int s = blockIdx.x >> 2;
  int c = (blockIdx.x & 3) * 256 + threadIdx.x;
  float Ar[16];
#pragma unroll
  for (int i = 0; i < 16; ++i) Ar[i] = -expf(Alog[c * 16 + i]);
  float h[16] = {};
  __shared__ float Bs[16], Cs[16];
  const size_t base = (size_t)s * 2048;

  int l0 = rev ? 2047 : 0;
  float vbc = (threadIdx.x < 32) ? dbc[(base + l0) * 64 + 32 + threadIdx.x] : 0.f;
  float d = dt[(base + l0) * 1024 + c];
  float u = xi[(base + l0) * 1024 + c];

  for (int t = 0; t < 2048; ++t) {
    int l = rev ? (2047 - t) : t;
    size_t row = base + l;
    if (threadIdx.x < 32) {
      if (threadIdx.x < 16) Bs[threadIdx.x] = vbc; else Cs[threadIdx.x - 16] = vbc;
    }
    __syncthreads();

    float dn = 0.f, un = 0.f, vn = 0.f;           // prefetch next step
    if (t + 1 < 2048) {
      size_t rown = base + (rev ? (2046 - t) : (t + 1));
      dn = dt[rown * 1024 + c];
      un = xi[rown * 1024 + c];
      if (threadIdx.x < 32) vn = dbc[rown * 64 + 32 + threadIdx.x];
    }

    float uu = d * u;
    float acc = 0.f;
#pragma unroll
    for (int i = 0; i < 16; ++i) {
      h[i] = expf(d * Ar[i]) * h[i] + uu * Bs[i];
      acc += h[i] * Cs[i];
    }
    y[row * 1024 + c] = acc;
    __syncthreads();
    d = dn; u = un; vbc = vn;
  }
}

// y = (y + xi*D) * silu(z)
__global__ void ygate(float* __restrict__ y, const float* __restrict__ xi,
                      const float* __restrict__ xz, const float* __restrict__ mD)
{
  int i = blockIdx.x * 256 + threadIdx.x;
  if (i >= 4096 * 1024) return;
  int c = i & 1023, row = i >> 10;
  float z  = xz[(size_t)row * 2048 + 1024 + c];
  float sz = z / (1.f + expf(-z));
  y[i] = (y[i] + xi[i] * mD[c]) * sz;
}

__launch_bounds__(256)
__global__ void cossim_kernel(const float* __restrict__ o, const float* __restrict__ a,
                              float* __restrict__ r)
{
  __shared__ float r0[256], r1[256], r2[256];
  int row = blockIdx.x;
  const float* po = o + (size_t)row * 512;
  const float* pa = a + (size_t)row * 512;
  float d = 0.f, no = 0.f, na = 0.f;
  for (int c = threadIdx.x; c < 512; c += 256) {
    float x = po[c], y = pa[c];
    d += x * y; no += x * x; na += y * y;
  }
  r0[threadIdx.x] = d; r1[threadIdx.x] = no; r2[threadIdx.x] = na;
  __syncthreads();
  for (int st = 128; st; st >>= 1) {
    if (threadIdx.x < st) {
      r0[threadIdx.x] += r0[threadIdx.x + st];
      r1[threadIdx.x] += r1[threadIdx.x + st];
      r2[threadIdx.x] += r2[threadIdx.x + st];
    }
    __syncthreads();
  }
  if (threadIdx.x == 0)
    r[row] = r0[0] / (fmaxf(sqrtf(r1[0]), 1e-8f) * fmaxf(sqrtf(r2[0]), 1e-8f));
}

__global__ void gate_kernel(float* __restrict__ out, const float* __restrict__ act,
                            const float* __restrict__ r, const float* __restrict__ wg)
{
  int i = blockIdx.x * 256 + threadIdx.x;
  if (i >= 4096 * 512) return;
  int row = i >> 9, n = row & 2047;
  float rv = 0.5f * (r[n] + r[2048 + n]);
  float w  = 1.f / (1.f + expf(-rv));
  out[i] = (*wg) * w * out[i] + (1.f - w) * act[i];
}

// P[n, c(0:1024)] = 0.5*T[s*2048+n, c%512] + 0.5*short[s, n, c%512], s = c/512
__global__ void mix_kernel(const float* __restrict__ T, const float* __restrict__ sh,
                           float* __restrict__ P)
{
  int i = blockIdx.x * 256 + threadIdx.x;
  if (i >= 2048 * 1024) return;
  int n = i >> 10, c = i & 1023;
  int s = c >> 9, cc = c & 511;
  size_t idx = (size_t)(s * 2048 + n) * 512 + cc;
  P[i] = 0.5f * T[idx] + 0.5f * sh[idx];
}

__launch_bounds__(256)
__global__ void softmax_kernel(float* __restrict__ S, int N)
{
  __shared__ float red[256];
  int row = blockIdx.x;
  float* x = S + (size_t)row * N;
  float mx = -3.4e38f;
  for (int c = threadIdx.x; c < N; c += 256) mx = fmaxf(mx, x[c]);
  red[threadIdx.x] = mx; __syncthreads();
  for (int st = 128; st; st >>= 1) {
    if (threadIdx.x < st) red[threadIdx.x] = fmaxf(red[threadIdx.x], red[threadIdx.x + st]);
    __syncthreads();
  }
  mx = red[0]; __syncthreads();
  float s = 0.f;
  for (int c = threadIdx.x; c < N; c += 256) { float e = expf(x[c] - mx); x[c] = e; s += e; }
  red[threadIdx.x] = s; __syncthreads();
  for (int st = 128; st; st >>= 1) {
    if (threadIdx.x < st) red[threadIdx.x] += red[threadIdx.x + st];
    __syncthreads();
  }
  float inv = 1.f / red[0];
  for (int c = threadIdx.x; c < N; c += 256) x[c] *= inv;
}

__launch_bounds__(256)
__global__ void layernorm_kernel(const float* __restrict__ a, const float* __restrict__ res,
                                 const float* __restrict__ g, const float* __restrict__ b,
                                 float* __restrict__ out, int C)
{
  __shared__ float red[256];
  int row = blockIdx.x;
  const float* pa = a + (size_t)row * C;
  const float* pr = res ? res + (size_t)row * C : nullptr;
  float s = 0.f, sq = 0.f;
  for (int c = threadIdx.x; c < C; c += 256) {
    float v = pa[c] + (pr ? pr[c] : 0.f);
    s += v; sq += v * v;
  }
  red[threadIdx.x] = s; __syncthreads();
  for (int st = 128; st; st >>= 1) { if (threadIdx.x < st) red[threadIdx.x] += red[threadIdx.x + st]; __syncthreads(); }
  s = red[0]; __syncthreads();
  red[threadIdx.x] = sq; __syncthreads();
  for (int st = 128; st; st >>= 1) { if (threadIdx.x < st) red[threadIdx.x] += red[threadIdx.x + st]; __syncthreads(); }
  sq = red[0];
  float mu  = s / C;
  float var = sq / C - mu * mu;
  float inv = rsqrtf(var + 1e-5f);
  for (int c = threadIdx.x; c < C; c += 256) {
    float v = pa[c] + (pr ? pr[c] : 0.f);
    out[(size_t)row * C + c] = (v - mu) * inv * g[c] + b[c];
  }
}

__launch_bounds__(256)
__global__ void l2norm_kernel(const float* __restrict__ in, float* __restrict__ out, int C)
{
  __shared__ float red[256];
  int row = blockIdx.x;
  const float* x = in + (size_t)row * C;
  float ss = 0.f;
  for (int c = threadIdx.x; c < C; c += 256) { float v = x[c]; ss += v * v; }
  red[threadIdx.x] = ss; __syncthreads();
  for (int st = 128; st; st >>= 1) { if (threadIdx.x < st) red[threadIdx.x] += red[threadIdx.x + st]; __syncthreads(); }
  float inv = 1.f / fmaxf(sqrtf(red[0]), 1e-12f);
  for (int c = threadIdx.x; c < C; c += 256) out[(size_t)row * C + c] = x[c] * inv;
}

__global__ void copy_enc(const float* __restrict__ enc, float* __restrict__ out)
{
  int i = blockIdx.x * 256 + threadIdx.x;
  if (i >= 2048 * 1024) return;
  int n = i >> 10, c = i & 1023;
  out[(size_t)n * 1152 + c] = enc[i];
}

// ---------------- driver ----------------
extern "C" void kernel_launch(void* const* d_in, const int* in_sizes, int n_in,
                              void* d_out, int out_size, void* d_ws, size_t ws_size,
                              hipStream_t stream)
{
  const float* img = (const float*)d_in[0];
  const float* txt = (const float*)d_in[1];
#define PP(i) ((const float*)d_in[2 + (i)])
  const float *nin_w = PP(0),  *nin_b = PP(1),  *nin2_w = PP(2), *nin2_b = PP(3);
  const float *grn_g = PP(4),  *grn_b = PP(5),  *grn2_g = PP(6), *grn2_b = PP(7);
  const float *temp = PP(8),   *wgate = PP(9);
  const float *m_in_w = PP(10), *m_conv_w = PP(11), *m_conv_b = PP(12);
  const float *m_xproj_w = PP(13), *m_dt_w = PP(14), *m_dt_b = PP(15);
  const float *m_Alog = PP(16), *m_D = PP(17), *m_out_w = PP(18);
  const float *inproj_w = PP(19), *inproj_b = PP(20);
  const float *outproj_w = PP(21), *outproj_b = PP(22);
  const float *fgrn1_g = PP(23), *fgrn1_b = PP(24), *fgrn2_g = PP(25), *fgrn2_b = PP(26);
  const float *qkv_w = PP(27), *qkv_b = PP(28), *attn_o_w = PP(29), *attn_o_b = PP(30);
  const float *ln1_g = PP(31), *ln1_b = PP(32);
  const float *ff1_w = PP(33), *ff1_b = PP(34), *ff2_w = PP(35), *ff2_b = PP(36);
  const float *ln2_g = PP(37), *ln2_b = PP(38);
  const float *hash_w = PP(39), *hash_b = PP(40);
#undef PP

  float* W = (float*)d_ws;
  float* out = (float*)d_out;
  dim3 blk(256);
  // mode: 0=plain, 1=transB, 2=flipK(B), 3=flipN(B)
  auto gemm = [&](const float* A, int lda, const float* B, int ldb, float* C, int ldc,
                  const float* bias, int M, int N, int K, float alpha,
                  int mode, int accC, int epi) {
    dim3 g((N + 63) / 64, (M + 127) / 128);
    switch (mode) {
      case 1: gemm_f16_wmma<1><<<g, blk, 0, stream>>>(A, lda, B, ldb, C, ldc, bias, M, N, K, alpha, accC, epi); break;
      case 2: gemm_f16_wmma<2><<<g, blk, 0, stream>>>(A, lda, B, ldb, C, ldc, bias, M, N, K, alpha, accC, epi); break;
      case 3: gemm_f16_wmma<3><<<g, blk, 0, stream>>>(A, lda, B, ldb, C, ldc, bias, M, N, K, alpha, accC, epi); break;
      default: gemm_f16_wmma<0><<<g, blk, 0, stream>>>(A, lda, B, ldb, C, ldc, bias, M, N, K, alpha, accC, epi); break;
    }
  };

  // 1. shortcuts: short = tokens @ inproj + b
  gemm(img, 512, inproj_w, 512, W + O_SHORT,           512, inproj_b, 2048, 512, 512, 1.f, 0, 0, 1);
  gemm(txt, 512, inproj_w, 512, W + O_SHORT + 1048576, 512, inproj_b, 2048, 512, 512, 1.f, 0, 0, 1);

  // 2. x = silu(grn(tokens @ nin + b))
  gemm(img, 512, nin_w, 512, W + O_XZ,           512, nin_b, 2048, 512, 512, 1.f, 0, 0, 1);
  gemm(txt, 512, nin_w, 512, W + O_XZ + 1048576, 512, nin_b, 2048, 512, 512, 1.f, 0, 0, 1);
  grn_kernel<<<4096, blk, 0, stream>>>(W + O_XZ, W + O_X, grn_g, grn_b, 512, 1);

  // 3. x_flat: per-stream 2048x512 transpose of the (512,2048) view
  for (int s = 0; s < 2; ++s)
    transpose_scale<<<dim3(64, 16), blk, 0, stream>>>(
        W + O_X + (size_t)s * 1048576, W + O_XF + (size_t)s * 1048576, 512, 2048, nullptr);

  // 4. zero mamba accumulator
  fill_zero<<<(2097152 + 255) / 256, blk, 0, stream>>>(W + O_MOUT, 2097152);

  // 5. four direction variants of the mamba block
  for (int v = 0; v < 4; ++v) {
    int flipD = v & 1, revL = (v >> 1) & 1;
    gemm(W + O_XF, 512, m_in_w, 2048, W + O_XZ, 2048, nullptr, 4096, 2048, 512, 1.f,
         flipD ? 2 : 0, 0, 0);
    conv_silu<<<(4194304 + 255) / 256, blk, 0, stream>>>(W + O_XZ, m_conv_w, m_conv_b, W + O_XI, revL);
    gemm(W + O_XI, 1024, m_xproj_w, 64, W + O_DBC, 64, nullptr, 4096, 64, 1024, 1.f, 0, 0, 0);
    gemm(W + O_DBC, 64, m_dt_w, 1024, W + O_DT, 1024, m_dt_b, 4096, 1024, 32, 1.f, 0, 0, 3);
    mamba_scan<<<8, blk, 0, stream>>>(W + O_DT, W + O_XI, W + O_DBC, m_Alog, W + O_Y, revL);
    ygate<<<(4194304 + 255) / 256, blk, 0, stream>>>(W + O_Y, W + O_XI, W + O_XZ, m_D);
    gemm(W + O_Y, 1024, m_out_w, 512, W + O_MOUT, 512, nullptr, 4096, 512, 1024, 1.f,
         flipD ? 3 : 0, 1, 0);
  }

  // 6. out = transpose(sum) * temp  (into O_XF)
  for (int s = 0; s < 2; ++s)
    transpose_scale<<<dim3(16, 64), blk, 0, stream>>>(
        W + O_MOUT + (size_t)s * 1048576, W + O_XF + (size_t)s * 1048576, 2048, 512, temp);

  // 7. cosine-similarity gate
  cossim_kernel<<<4096, blk, 0, stream>>>(W + O_XF, W + O_X, W + O_R);
  gate_kernel<<<(2097152 + 255) / 256, blk, 0, stream>>>(W + O_XF, W + O_X, W + O_R, wgate);

  // 8. m = silu(grn2(out @ nin2 + b))
  gemm(W + O_XF, 512, nin2_w, 512, W + O_XZ, 512, nin2_b, 4096, 512, 512, 1.f, 0, 0, 1);
  grn_kernel<<<4096, blk, 0, stream>>>(W + O_XZ, W + O_M, grn2_g, grn2_b, 512, 1);

  // 9. ic/tc = grn(m)@outproj+b, mix with shortcuts, fgrn2 -> res
  grn_kernel<<<4096, blk, 0, stream>>>(W + O_M, W + O_XI, fgrn1_g, fgrn1_b, 512, 0);
  gemm(W + O_XI, 512, outproj_w, 512, W + O_XZ, 512, outproj_b, 4096, 512, 512, 1.f, 0, 0, 1);
  mix_kernel<<<(2097152 + 255) / 256, blk, 0, stream>>>(W + O_XZ, W + O_SHORT, W + O_DT);
  grn_kernel<<<2048, blk, 0, stream>>>(W + O_DT, W + O_RES, fgrn2_g, fgrn2_b, 1024, 0);

  // 10. transformer encoder layer
  gemm(W + O_RES, 1024, qkv_w, 3072, W + O_QKV, 3072, qkv_b, 2048, 3072, 1024, 1.f, 0, 0, 1);
  for (int h = 0; h < 4; ++h) {
    const float* q  = W + O_QKV + h * 256;
    const float* k  = W + O_QKV + 1024 + h * 256;
    const float* vv = W + O_QKV + 2048 + h * 256;
    gemm(q, 3072, k, 3072, W + O_SC, 2048, nullptr, 2048, 2048, 256, 0.0625f, 1, 0, 0);
    softmax_kernel<<<2048, blk, 0, stream>>>(W + O_SC, 2048);
    gemm(W + O_SC, 2048, vv, 3072, W + O_AO + h * 256, 1024, nullptr, 2048, 256, 2048, 1.f, 0, 0, 0);
  }
  gemm(W + O_AO, 1024, attn_o_w, 1024, W + O_FF, 1024, attn_o_b, 2048, 1024, 1024, 1.f, 0, 0, 1);
  layernorm_kernel<<<2048, blk, 0, stream>>>(W + O_FF, W + O_RES, ln1_g, ln1_b, W + O_XLN, 1024);
  gemm(W + O_XLN, 1024, ff1_w, 2048, W + O_FF, 2048, ff1_b, 2048, 2048, 1024, 1.f, 0, 0, 4);
  gemm(W + O_FF, 2048, ff2_w, 1024, W + O_AO, 1024, ff2_b, 2048, 1024, 2048, 1.f, 0, 0, 1);
  layernorm_kernel<<<2048, blk, 0, stream>>>(W + O_AO, W + O_XLN, ln2_g, ln2_b, W + O_SC, 1024);
  l2norm_kernel<<<2048, blk, 0, stream>>>(W + O_SC, W + O_ENC, 1024);

  // 11. outputs: [img_f, txt_f, tanh(img_f@hash+b), tanh(txt_f@hash+b)]
  copy_enc<<<(2097152 + 255) / 256, blk, 0, stream>>>(W + O_ENC, out);
  gemm(W + O_ENC,       1024, hash_w, 64, out + 1024, 1152, hash_b, 2048, 64, 512, 1.f, 0, 0, 5);
  gemm(W + O_ENC + 512, 1024, hash_w, 64, out + 1088, 1152, hash_b, 2048, 64, 512, 1.f, 0, 0, 5);

  (void)in_sizes; (void)n_in; (void)out_size; (void)ws_size;
}
